// StudentModel_70205535420685
// MI455X (gfx1250) — matrix operand order repaired and checked by reading, not compiled, not run on hardware
//
#include <hip/hip_runtime.h>
#include <hip/hip_bf16.h>
#include <math.h>
#include <stdint.h>

// ---------------------------------------------------------------------------
// Model dims
// ---------------------------------------------------------------------------
#define BDIM   64
#define SDIM   512
#define MROWS  (BDIM * SDIM)   // 32768 token rows
#define DMODEL 384
#define NHEAD  12
#define DK     32
#define NLAYER 6
#define FFDIM  1536

typedef __attribute__((ext_vector_type(16))) _Float16 v16h;
typedef __attribute__((ext_vector_type(8)))  _Float16 v8h;
typedef __attribute__((ext_vector_type(8)))  float    v8f;

union H16 { v16h v; v8h h[2]; };

__device__ __forceinline__ float sigf(float x) { return 1.0f / (1.0f + __expf(-x)); }

// ---------------------------------------------------------------------------
// CDNA5 async global->LDS copy, GVS addressing: mem = SGPR64 + VGPR_I32.
// The wave-uniform base (incl. K-step advance) rides in SGPRs; the per-lane
// offset VGPR is constant across the whole K loop. ASYNCcnt-tracked.
// LDS byte address = low 32 bits of the generic shared-aperture address.
// ---------------------------------------------------------------------------
__device__ __forceinline__ void async_copy16_s(uint32_t lds_dst, uint32_t voff,
                                               const void* sbase)
{
    asm volatile("global_load_async_to_lds_b128 %0, %1, %2"
                 :: "v"(lds_dst), "v"(voff), "s"(sbase) : "memory");
}
__device__ __forceinline__ void wait_async0()
{
    asm volatile("s_wait_asynccnt 0x0" ::: "memory");
}

// ---------------------------------------------------------------------------
// Fused embedding + LayerNorm: one wave per token row.
// ---------------------------------------------------------------------------
__global__ __launch_bounds__(256)
void embed_ln_kernel(const int* __restrict__ ids,
                     const float* __restrict__ wemb,
                     const float* __restrict__ pemb,
                     const float* __restrict__ temb,
                     const float* __restrict__ lnw,
                     const float* __restrict__ lnb,
                     float* __restrict__ x32,
                     _Float16* __restrict__ x16)
{
    int row  = blockIdx.x * 8 + (threadIdx.x >> 5);
    int lane = threadIdx.x & 31;
    int s    = row & (SDIM - 1);
    int id   = ids[row];
    const float* wr = wemb + (size_t)id * DMODEL;
    const float* pr = pemb + (size_t)s * DMODEL;

    float v[12];
    float sum = 0.f;
#pragma unroll
    for (int i = 0; i < 12; i++) {
        int d = lane + 32 * i;
        v[i] = wr[d] + pr[d] + temb[d];
        sum += v[i];
    }
#pragma unroll
    for (int off = 16; off > 0; off >>= 1) sum += __shfl_xor(sum, off, 32);
    float mean = sum * (1.0f / DMODEL);
    float var = 0.f;
#pragma unroll
    for (int i = 0; i < 12; i++) { float d = v[i] - mean; var += d * d; }
#pragma unroll
    for (int off = 16; off > 0; off >>= 1) var += __shfl_xor(var, off, 32);
    float r = rsqrtf(var * (1.0f / DMODEL) + 1e-12f);

    size_t base = (size_t)row * DMODEL;
#pragma unroll
    for (int i = 0; i < 12; i++) {
        int d = lane + 32 * i;
        float y = (v[i] - mean) * r * lnw[d] + lnb[d];
        x32[base + d] = y;
        x16[base + d] = (_Float16)y;
    }
}

// ---------------------------------------------------------------------------
// LayerNorm kernel (wave per row), reads pre-LN f32, writes f32 + f16.
// ---------------------------------------------------------------------------
__global__ __launch_bounds__(256)
void ln_kernel(const float* __restrict__ in,
               const float* __restrict__ lnw,
               const float* __restrict__ lnb,
               float* __restrict__ x32,
               _Float16* __restrict__ x16)
{
    int row  = blockIdx.x * 8 + (threadIdx.x >> 5);
    int lane = threadIdx.x & 31;
    const float* xr = in + (size_t)row * DMODEL;
    float v[12];
    float sum = 0.f;
#pragma unroll
    for (int i = 0; i < 12; i++) { v[i] = xr[lane + 32 * i]; sum += v[i]; }
#pragma unroll
    for (int off = 16; off > 0; off >>= 1) sum += __shfl_xor(sum, off, 32);
    float mean = sum * (1.0f / DMODEL);
    float var = 0.f;
#pragma unroll
    for (int i = 0; i < 12; i++) { float d = v[i] - mean; var += d * d; }
#pragma unroll
    for (int off = 16; off > 0; off >>= 1) var += __shfl_xor(var, off, 32);
    float r = rsqrtf(var * (1.0f / DMODEL) + 1e-12f);
    size_t base = (size_t)row * DMODEL;
#pragma unroll
    for (int i = 0; i < 12; i++) {
        int d = lane + 32 * i;
        float y = (v[i] - mean) * r * lnw[d] + lnb[d];
        x32[base + d] = y;
        x16[base + d] = (_Float16)y;
    }
}

// ---------------------------------------------------------------------------
// Weight convert: W f32 [K,N] row-major -> BT f16 [N,K] (transposed).
// ---------------------------------------------------------------------------
__global__ __launch_bounds__(256)
void convert_wt_kernel(const float* __restrict__ W, _Float16* __restrict__ BT,
                       int K, int N)
{
    int idx = blockIdx.x * 256 + threadIdx.x;
    if (idx >= K * N) return;
    int k = idx / N, n = idx - k * N;
    BT[(size_t)n * K + k] = (_Float16)W[idx];
}

// ---------------------------------------------------------------------------
// WMMA GEMM:  out[M,N] = A[M,K](f16) * BT[N,K]^T (f16) + bias (+res) (gelu?)
// Block tile 128x128, 8 waves (4Mx2N), wave tile 32x64 = 2x4 fragments,
// K-step 32 via v_wmma_f32_16x16x32_f16.
// Staging: double-buffered LDS filled by async copies in GVS mode; the
// K-advance is scalar (SGPR base), per-lane offsets are 2 fixed VGPRs.
// Copy of tile k+1 overlaps the 8 WMMAs of tile k; one barrier per K-step.
// ---------------------------------------------------------------------------
template <bool GELU, bool RES, bool OUTF, bool OUTH, int N, int K>
__global__ __launch_bounds__(256)
void gemm_kernel(const _Float16* __restrict__ A,
                 const _Float16* __restrict__ BT,
                 const float* __restrict__ bias,
                 const float* __restrict__ res,
                 float* __restrict__ outF,
                 _Float16* __restrict__ outH)
{
    // +8-half pad => row stride 80 B (16B-aligned chunks, bank-spread)
    __shared__ _Float16 As[2][128][40];
    __shared__ _Float16 Bs[2][128][40];

    const int m0   = blockIdx.y * 128;
    const int n0   = blockIdx.x * 128;
    const int tid  = threadIdx.x;
    const int wid  = tid >> 5;
    const int lane = tid & 31;
    const int waveM = wid & 3;    // 4 waves over M
    const int waveN = wid >> 2;   // 2 waves over N
    const int hs  = lane >> 4;
    const int l16 = lane & 15;

    // each thread stages 2 A-chunks + 2 B-chunks (16B each) per K-step
    const int ldRow0 = tid >> 2;          // 0..63
    const int ldRow1 = (tid + 256) >> 2;  // 64..127
    const int ldPart = (tid & 3) * 8;     // half offset within row

    // per-lane byte offsets within a tile's rows: constant for all K-steps,
    // shared by the A and B tiles (same row geometry)
    const uint32_t of0 = (uint32_t)((ldRow0 * K + ldPart) * sizeof(_Float16));
    const uint32_t of1 = (uint32_t)((ldRow1 * K + ldPart) * sizeof(_Float16));

    // wave-uniform bases (go to SGPRs); K-step advance is scalar arithmetic
    const char* baseA = (const char*)(A  + (size_t)m0 * K);
    const char* baseB = (const char*)(BT + (size_t)n0 * K);

    // fixed LDS destination addresses (2 buffers x {A0,A1,B0,B1})
    const uint32_t lA0[2] = { (uint32_t)(uintptr_t)&As[0][ldRow0][ldPart],
                              (uint32_t)(uintptr_t)&As[1][ldRow0][ldPart] };
    const uint32_t lA1[2] = { (uint32_t)(uintptr_t)&As[0][ldRow1][ldPart],
                              (uint32_t)(uintptr_t)&As[1][ldRow1][ldPart] };
    const uint32_t lB0[2] = { (uint32_t)(uintptr_t)&Bs[0][ldRow0][ldPart],
                              (uint32_t)(uintptr_t)&Bs[1][ldRow0][ldPart] };
    const uint32_t lB1[2] = { (uint32_t)(uintptr_t)&Bs[0][ldRow1][ldPart],
                              (uint32_t)(uintptr_t)&Bs[1][ldRow1][ldPart] };

    v8f acc[2][4] = {};

    constexpr int nsteps = K >> 5;

    auto stage = [&](int buf, int kstep) {
        const char* sA = baseA + (size_t)kstep * 64;   // 32 halfs = 64 B
        const char* sB = baseB + (size_t)kstep * 64;
        async_copy16_s(lA0[buf], of0, sA);
        async_copy16_s(lA1[buf], of1, sA);
        async_copy16_s(lB0[buf], of0, sB);
        async_copy16_s(lB1[buf], of1, sB);
    };

    auto compute_step = [&](int cur) {
        H16 afr[2], bfr[4];
#pragma unroll
        for (int mt = 0; mt < 2; mt++) {
            int m = waveM * 32 + mt * 16 + l16;
            afr[mt].h[0] = *(const v8h*)&As[cur][m][hs * 8];
            afr[mt].h[1] = *(const v8h*)&As[cur][m][16 + hs * 8];
        }
#pragma unroll
        for (int nt = 0; nt < 4; nt++) {
            int n = waveN * 64 + nt * 16 + l16;
            bfr[nt].h[0] = *(const v8h*)&Bs[cur][n][hs * 8];
            bfr[nt].h[1] = *(const v8h*)&Bs[cur][n][16 + hs * 8];
        }
#pragma unroll
        for (int mt = 0; mt < 2; mt++)
#pragma unroll
            for (int nt = 0; nt < 4; nt++)
                acc[mt][nt] = __builtin_amdgcn_wmma_f32_16x16x32_f16(
                    false, afr[mt].v, false, bfr[nt].v,
                    (short)0, acc[mt][nt], false, false);
    };

    // prologue: stage tile 0 into buffer 0
    stage(0, 0);
    wait_async0();
    __syncthreads();

    // steady state: prefetch tile step+1 while computing tile step
#pragma unroll
    for (int step = 0; step < nsteps - 1; step++) {
        stage((step + 1) & 1, step + 1);
        compute_step(step & 1);
        wait_async0();   // next tile resident (this wave's copies)
        __syncthreads(); // all waves done reading cur + next visible
    }
    // last step (no prefetch, no trailing barrier)
    compute_step((nsteps - 1) & 1);

    // epilogue: immediate-offset stores (r*N is compile-time)
#pragma unroll
    for (int mt = 0; mt < 2; mt++) {
#pragma unroll
        for (int nt = 0; nt < 4; nt++) {
            const int n = n0 + waveN * 64 + nt * 16 + l16;
            const int mBase = m0 + waveM * 32 + mt * 16 + hs * 8;
            const float bn = bias[n];
            const size_t o0 = (size_t)mBase * N + n;
            float*       pF = OUTF ? outF + o0 : nullptr;
            _Float16*    pH = OUTH ? outH + o0 : nullptr;
            const float* pR = RES  ? res  + o0 : nullptr;
#pragma unroll
            for (int r = 0; r < 8; r++) {
                float v = acc[mt][nt][r] + bn;
                if (RES) v += pR[r * N];
                if (GELU) v = 0.5f * v * (1.0f + erff(v * 0.70710678118654752f));
                if (OUTF) pF[r * N] = v;
                if (OUTH) pH[r * N] = (_Float16)v;
            }
        }
    }
}

// ---------------------------------------------------------------------------
// beta / g projections (D -> H=12): one thread per (row, head).
// ---------------------------------------------------------------------------
__global__ __launch_bounds__(256)
void betag_kernel(const float* __restrict__ x,
                  const float* __restrict__ Wb, const float* __restrict__ bb,
                  const float* __restrict__ Wg, const float* __restrict__ bg,
                  const int* __restrict__ mask,
                  float* __restrict__ beta, float* __restrict__ g)
{
    int idx = blockIdx.x * 256 + threadIdx.x;      // row * 12 + h
    if (idx >= MROWS * NHEAD) return;
    int row = idx / NHEAD, h = idx - row * NHEAD;
    const float* xr = x + (size_t)row * DMODEL;
    float sb = bb[h], sg = bg[h];
    for (int d = 0; d < DMODEL; d++) {
        float xv = xr[d];
        sb = fmaf(xv, Wb[d * NHEAD + h], sb);
        sg = fmaf(xv, Wg[d * NHEAD + h], sg);
    }
    float m = (float)mask[row];
    beta[idx] = m * sigf(sb);
    g[idx]    = sigf(sg);
}

// ---------------------------------------------------------------------------
// DeltaNet recurrence: one wave per (b,h) chain, lane = d (DK==32==wave32).
// Both 32x32 states live entirely in VGPRs; k_t/q_t broadcast via __shfl.
// ---------------------------------------------------------------------------
__global__ __launch_bounds__(32)
void deltanet_kernel(const float* __restrict__ q,
                     const float* __restrict__ k,
                     const float* __restrict__ v,
                     const float* __restrict__ beta,
                     const float* __restrict__ g,
                     const float* __restrict__ fl,
                     const float* __restrict__ sl,
                     _Float16* __restrict__ out16)
{
    int bh = blockIdx.x;
    int b = bh / NHEAD, h = bh - b * NHEAD;
    int lane = threadIdx.x;
    float af  = sigf(fl[h]);
    float as_ = sigf(sl[h]);

    float Sf[32], Ss[32];
#pragma unroll
    for (int j = 0; j < 32; j++) { Sf[j] = 0.f; Ss[j] = 0.f; }

    for (int s = 0; s < SDIM; s++) {
        size_t base = ((size_t)(b * SDIM + s)) * DMODEL + h * DK;
        float qt = q[base + lane];
        float kt = k[base + lane];
        float vt = v[base + lane];

        float nr = kt * kt;
#pragma unroll
        for (int off = 16; off > 0; off >>= 1) nr += __shfl_xor(nr, off, 32);
        kt = kt / (sqrtf(nr) + 1e-6f);

        int bgi = (b * SDIM + s) * NHEAD + h;
        float bt = beta[bgi];
        float gt = g[bgi];

        float ktb[32], qtb[32];
#pragma unroll
        for (int j = 0; j < 32; j++) {
            ktb[j] = __shfl(kt, j, 32);
            qtb[j] = __shfl(qt, j, 32);
        }

        float rf = 0.f, rs = 0.f;
#pragma unroll
        for (int j = 0; j < 32; j++) {
            rf = fmaf(Sf[j], ktb[j], rf);
            rs = fmaf(Ss[j], ktb[j], rs);
        }
        float df  = bt * (vt - rf);
        float ds_ = bt * (vt - rs);

        float of = 0.f, os = 0.f;
#pragma unroll
        for (int j = 0; j < 32; j++) {
            Sf[j] = fmaf(af,  Sf[j], ktb[j] * df);
            of    = fmaf(Sf[j], qtb[j], of);
            Ss[j] = fmaf(as_, Ss[j], ktb[j] * ds_);
            os    = fmaf(Ss[j], qtb[j], os);
        }
        float o = gt * of + (1.0f - gt) * os;
        out16[base + lane] = (_Float16)o;
    }
}

// ---------------------------------------------------------------------------
// Masked mean-pool over S + L2 normalize -> d_out [B, D] f32.
// ---------------------------------------------------------------------------
__global__ __launch_bounds__(128)
void pool_kernel(const float* __restrict__ x, const int* __restrict__ mask,
                 float* __restrict__ out)
{
    int b = blockIdx.x, tid = threadIdx.x;
    float acc[3] = {0.f, 0.f, 0.f};
    float cnt = 0.f;
    for (int s = 0; s < SDIM; s++) {
        float mm = (float)mask[b * SDIM + s];
        cnt += mm;
        const float* xr = x + ((size_t)(b * SDIM + s)) * DMODEL;
#pragma unroll
        for (int j = 0; j < 3; j++) acc[j] = fmaf(xr[tid + 128 * j], mm, acc[j]);
    }
    cnt = fmaxf(cnt, 1e-9f);
    float e[3]; float ss = 0.f;
#pragma unroll
    for (int j = 0; j < 3; j++) { e[j] = acc[j] / cnt; ss += e[j] * e[j]; }

    __shared__ float red[128];
    red[tid] = ss;
    __syncthreads();
    for (int off = 64; off > 0; off >>= 1) {
        if (tid < off) red[tid] += red[tid + off];
        __syncthreads();
    }
    float inv = rsqrtf(red[0]);
#pragma unroll
    for (int j = 0; j < 3; j++) out[(size_t)b * DMODEL + tid + 128 * j] = e[j] * inv;
}

// ---------------------------------------------------------------------------
// Host driver
// ---------------------------------------------------------------------------
extern "C" void kernel_launch(void* const* d_in, const int* in_sizes, int n_in,
                              void* d_out, int out_size, void* d_ws, size_t ws_size,
                              hipStream_t stream)
{
    (void)in_sizes; (void)n_in; (void)out_size; (void)ws_size;

    const int*   ids     = (const int*)  d_in[0];
    const int*   amask   = (const int*)  d_in[1];
    const float* wemb    = (const float*)d_in[2];
    const float* pemb    = (const float*)d_in[3];
    const float* temb    = (const float*)d_in[4];
    const float* elnw    = (const float*)d_in[5];
    const float* elnb    = (const float*)d_in[6];
    const float* Wq      = (const float*)d_in[7];
    const float* bq      = (const float*)d_in[8];
    const float* Wk      = (const float*)d_in[9];
    const float* bk      = (const float*)d_in[10];
    const float* Wv      = (const float*)d_in[11];
    const float* bv      = (const float*)d_in[12];
    const float* Wbeta   = (const float*)d_in[13];
    const float* bbeta   = (const float*)d_in[14];
    const float* Wg      = (const float*)d_in[15];
    const float* bg      = (const float*)d_in[16];
    const float* Wo      = (const float*)d_in[17];
    const float* bo      = (const float*)d_in[18];
    const float* flog    = (const float*)d_in[19];
    const float* slog    = (const float*)d_in[20];
    const float* alnw    = (const float*)d_in[21];
    const float* alnb    = (const float*)d_in[22];
    const float* W1      = (const float*)d_in[23];
    const float* b1      = (const float*)d_in[24];
    const float* W2      = (const float*)d_in[25];
    const float* b2      = (const float*)d_in[26];
    const float* flnw    = (const float*)d_in[27];
    const float* flnb    = (const float*)d_in[28];

    // --- workspace carve-out (~390 MB) ---
    char* ws = (char*)d_ws;
    auto carve = [&](size_t bytes) -> void* {
        void* p = (void*)ws;
        ws += (bytes + 255) & ~(size_t)255;
        return p;
    };
    const size_t MD = (size_t)MROWS * DMODEL;
    float*    x32    = (float*)   carve(MD * 4);
    _Float16* x16    = (_Float16*)carve(MD * 2);
    float*    q32    = (float*)   carve(MD * 4);
    float*    k32    = (float*)   carve(MD * 4);
    float*    v32    = (float*)   carve(MD * 4);
    float*    betaB  = (float*)   carve((size_t)MROWS * NHEAD * 4);
    float*    gB     = (float*)   carve((size_t)MROWS * NHEAD * 4);
    _Float16* attn16 = (_Float16*)carve(MD * 2);
    float*    pre32  = (float*)   carve(MD * 4);
    _Float16* h16    = (_Float16*)carve((size_t)MROWS * FFDIM * 2);
    _Float16* wt16   = (_Float16*)carve((size_t)FFDIM * DMODEL * 2);

    float* outp = (float*)d_out;

    const dim3 blk256(256), blk128(128), blk32(32);
    const dim3 gRow(MROWS / 8);                    // wave-per-row kernels
    const dim3 gGemmD(DMODEL / 128, MROWS / 128);  // N=384 -> 3 x 256
    const dim3 gGemmF(FFDIM / 128,  MROWS / 128);  // N=1536 -> 12 x 256
    const int  cvtDD = (DMODEL * DMODEL + 255) / 256;
    const int  cvtDF = (DMODEL * FFDIM  + 255) / 256;

    hipLaunchKernelGGL(embed_ln_kernel, gRow, blk256, 0, stream,
                       ids, wemb, pemb, temb, elnw, elnb, x32, x16);

    for (int l = 0; l < NLAYER; l++) {
        const size_t wDD = (size_t)l * DMODEL * DMODEL;
        const size_t wDF = (size_t)l * DMODEL * FFDIM;

        // Q / K / V projections
        hipLaunchKernelGGL(convert_wt_kernel, dim3(cvtDD), blk256, 0, stream,
                           Wq + wDD, wt16, DMODEL, DMODEL);
        hipLaunchKernelGGL((gemm_kernel<false, false, true, false, DMODEL, DMODEL>),
                           gGemmD, blk256, 0, stream,
                           x16, wt16, bq + l * DMODEL, (const float*)nullptr,
                           q32, (_Float16*)nullptr);
        hipLaunchKernelGGL(convert_wt_kernel, dim3(cvtDD), blk256, 0, stream,
                           Wk + wDD, wt16, DMODEL, DMODEL);
        hipLaunchKernelGGL((gemm_kernel<false, false, true, false, DMODEL, DMODEL>),
                           gGemmD, blk256, 0, stream,
                           x16, wt16, bk + l * DMODEL, (const float*)nullptr,
                           k32, (_Float16*)nullptr);
        hipLaunchKernelGGL(convert_wt_kernel, dim3(cvtDD), blk256, 0, stream,
                           Wv + wDD, wt16, DMODEL, DMODEL);
        hipLaunchKernelGGL((gemm_kernel<false, false, true, false, DMODEL, DMODEL>),
                           gGemmD, blk256, 0, stream,
                           x16, wt16, bv + l * DMODEL, (const float*)nullptr,
                           v32, (_Float16*)nullptr);

        // beta / g
        hipLaunchKernelGGL(betag_kernel, dim3((MROWS * NHEAD + 255) / 256), blk256,
                           0, stream,
                           x32, Wbeta + (size_t)l * DMODEL * NHEAD, bbeta + l * NHEAD,
                           Wg + (size_t)l * DMODEL * NHEAD, bg + l * NHEAD,
                           amask, betaB, gB);

        // recurrence: 768 chains, one wave each
        hipLaunchKernelGGL(deltanet_kernel, dim3(BDIM * NHEAD), blk32, 0, stream,
                           q32, k32, v32, betaB, gB,
                           flog + l * NHEAD, slog + l * NHEAD, attn16);

        // output projection + residual, then LN -> x
        hipLaunchKernelGGL(convert_wt_kernel, dim3(cvtDD), blk256, 0, stream,
                           Wo + wDD, wt16, DMODEL, DMODEL);
        hipLaunchKernelGGL((gemm_kernel<false, true, true, false, DMODEL, DMODEL>),
                           gGemmD, blk256, 0, stream,
                           attn16, wt16, bo + l * DMODEL, x32,
                           pre32, (_Float16*)nullptr);
        hipLaunchKernelGGL(ln_kernel, gRow, blk256, 0, stream,
                           pre32, alnw + l * DMODEL, alnb + l * DMODEL, x32, x16);

        // FFN
        hipLaunchKernelGGL(convert_wt_kernel, dim3(cvtDF), blk256, 0, stream,
                           W1 + wDF, wt16, DMODEL, FFDIM);
        hipLaunchKernelGGL((gemm_kernel<true, false, false, true, FFDIM, DMODEL>),
                           gGemmF, blk256, 0, stream,
                           x16, wt16, b1 + (size_t)l * FFDIM, (const float*)nullptr,
                           (float*)nullptr, h16);
        hipLaunchKernelGGL(convert_wt_kernel, dim3(cvtDF), blk256, 0, stream,
                           W2 + wDF, wt16, FFDIM, DMODEL);
        hipLaunchKernelGGL((gemm_kernel<false, true, true, false, DMODEL, FFDIM>),
                           gGemmD, blk256, 0, stream,
                           h16, wt16, b2 + l * DMODEL, x32,
                           pre32, (_Float16*)nullptr);
        hipLaunchKernelGGL(ln_kernel, gRow, blk256, 0, stream,
                           pre32, flnw + l * DMODEL, flnb + l * DMODEL, x32, x16);
    }

    hipLaunchKernelGGL(pool_kernel, dim3(BDIM), blk128, 0, stream,
                       x32, amask, outp);
}
